// BitLinear_77799037599823
// MI455X (gfx1250) — matrix operand order repaired and checked by reading, not compiled, not run on hardware
//
#include <hip/hip_runtime.h>
#include <cstdint>

typedef _Float16 v8h  __attribute__((ext_vector_type(8)));
typedef _Float16 v16h __attribute__((ext_vector_type(16)));
typedef float    v8f  __attribute__((ext_vector_type(8)));

#define IN_F  2048
#define OUT_F 2048
#define BATCH 8192
#define QB    7.0f
#define EPSF  1e-5f

// ----------------------------------------------------------------------------
// Kernel 1: weight prep. One block per output row.
// ----------------------------------------------------------------------------
__global__ void __launch_bounds__(256)
prep_weights_kernel(const float* __restrict__ w,
                    _Float16* __restrict__ wbin,
                    float* __restrict__ beta) {
    __shared__ float r1[256];
    __shared__ float r2[256];
    const int row = blockIdx.x;
    const int tid = threadIdx.x;
    const float* wr = w + (size_t)row * IN_F;

    float v[8];
    const float4* wv = (const float4*)(wr + tid * 8);
    float4 a = wv[0], b = wv[1];
    v[0]=a.x; v[1]=a.y; v[2]=a.z; v[3]=a.w;
    v[4]=b.x; v[5]=b.y; v[6]=b.z; v[7]=b.w;

    float s = 0.f, sa = 0.f;
#pragma unroll
    for (int i = 0; i < 8; ++i) { s += v[i]; sa += fabsf(v[i]); }
    r1[tid] = s; r2[tid] = sa;
    __syncthreads();
#pragma unroll
    for (int off = 128; off > 0; off >>= 1) {
        if (tid < off) { r1[tid] += r1[tid + off]; r2[tid] += r2[tid + off]; }
        __syncthreads();
    }
    const float mean = r1[0] * (1.0f / (float)IN_F);
    if (tid == 0) beta[row] = r2[0] * (1.0f / (float)IN_F);

    _Float16 o[8];
#pragma unroll
    for (int i = 0; i < 8; ++i) {
        float c = v[i] - mean;
        o[i] = (_Float16)((c > 0.f) ? 1.0f : ((c < 0.f) ? -1.0f : 0.0f));
    }
    *(v8h*)(wbin + (size_t)row * IN_F + tid * 8) = *(const v8h*)o;
}

// ----------------------------------------------------------------------------
// Kernel 2: activation prep. One block per batch row.
// ----------------------------------------------------------------------------
__global__ void __launch_bounds__(256)
prep_acts_kernel(const float* __restrict__ x,
                 _Float16* __restrict__ xq,
                 float* __restrict__ gama) {
    __shared__ float r1[256];
    __shared__ float r2[256];
    const int row = blockIdx.x;
    const int tid = threadIdx.x;
    const float* xr = x + (size_t)row * IN_F;

    float v[8];
    const float4* xv = (const float4*)(xr + tid * 8);
    float4 a = xv[0], b = xv[1];
    v[0]=a.x; v[1]=a.y; v[2]=a.z; v[3]=a.w;
    v[4]=b.x; v[5]=b.y; v[6]=b.z; v[7]=b.w;

    float s = 0.f, ss = 0.f;
#pragma unroll
    for (int i = 0; i < 8; ++i) { s += v[i]; ss += v[i] * v[i]; }
    r1[tid] = s; r2[tid] = ss;
    __syncthreads();
#pragma unroll
    for (int off = 128; off > 0; off >>= 1) {
        if (tid < off) { r1[tid] += r1[tid + off]; r2[tid] += r2[tid + off]; }
        __syncthreads();
    }
    const float mu  = r1[0] * (1.0f / (float)IN_F);
    const float var = r2[0] * (1.0f / (float)IN_F) - mu * mu;
    const float rs  = rsqrtf(var + EPSF);
    __syncthreads();

    float xl[8]; float am = 0.f;
#pragma unroll
    for (int i = 0; i < 8; ++i) { xl[i] = (v[i] - mu) * rs; am = fmaxf(am, fabsf(xl[i])); }
    r1[tid] = am;
    __syncthreads();
#pragma unroll
    for (int off = 128; off > 0; off >>= 1) {
        if (tid < off) r1[tid] = fmaxf(r1[tid], r1[tid + off]);
        __syncthreads();
    }
    const float g  = fmaxf(r1[0], EPSF);
    if (tid == 0) gama[row] = g;
    const float sc = QB / g;

    _Float16 o[8];
#pragma unroll
    for (int i = 0; i < 8; ++i) {
        float q = xl[i] * sc;
        q = fminf(fmaxf(q, -QB + EPSF), QB - EPSF);
        o[i] = (_Float16)q;
    }
    *(v8h*)(xq + (size_t)row * IN_F + tid * 8) = *(const v8h*)o;
}

// ----------------------------------------------------------------------------
// Kernel 3: WMMA GEMM, 3-stage async global->LDS pipeline (ASYNCcnt path,
// using CDNA5's 320KB LDS). Block tile 256x128, BK=32, 8 waves as 4x2, wave
// tile 64x64 => 16 v_wmma_f32_16x16x32_f16 per wave per K-step. Steady-state
// loop is branchless with a constant s_wait_asynccnt 0x6 (async loads retire
// in order, so <=6 outstanding means the batch consumed next step landed);
// the last two K-steps are peeled (drain wait 0x0, then no staging at all).
// ----------------------------------------------------------------------------
#define BM 256
#define BN 128
#define BK 32
#define SA 40                  // padded LDS row stride in halves (80 B)
#define NSTEP (IN_F / BK)      // 64 K-steps
#define KSTEP_BYTES (BK * 2)   // 64 B advance per step
#define NBUF 3

// 6 async b128 loads issued as one clause; s_wait_xcnt 0 protects the address
// VGPRs we bump right after (inline-asm loads are invisible to the compiler's
// own xcnt bookkeeping).
__device__ __forceinline__ void async_stage(uint32_t la, uint32_t lb,
                                            const uint32_t (&aL)[4], uint64_t (&aG)[4],
                                            const uint32_t (&bL)[2], uint64_t (&bG)[2]) {
    asm volatile(
        "s_clause 0x5\n\t"
        "global_load_async_to_lds_b128 %0, %6, off\n\t"
        "global_load_async_to_lds_b128 %1, %7, off\n\t"
        "global_load_async_to_lds_b128 %2, %8, off\n\t"
        "global_load_async_to_lds_b128 %3, %9, off\n\t"
        "global_load_async_to_lds_b128 %4, %10, off\n\t"
        "global_load_async_to_lds_b128 %5, %11, off\n\t"
        "s_wait_xcnt 0x0"
        :: "v"(la + aL[0]), "v"(la + aL[1]), "v"(la + aL[2]), "v"(la + aL[3]),
           "v"(lb + bL[0]), "v"(lb + bL[1]),
           "v"(aG[0]), "v"(aG[1]), "v"(aG[2]), "v"(aG[3]),
           "v"(bG[0]), "v"(bG[1])
        : "memory");
#pragma unroll
    for (int j = 0; j < 4; ++j) aG[j] += KSTEP_BYTES;
#pragma unroll
    for (int j = 0; j < 2; ++j) bG[j] += KSTEP_BYTES;
}

__global__ void __launch_bounds__(256)
bitlinear_gemm_kernel(const _Float16* __restrict__ Aq,
                      const _Float16* __restrict__ Bw,
                      const float* __restrict__ gama,
                      const float* __restrict__ beta,
                      const float* __restrict__ bias,
                      float* __restrict__ out) {
    __shared__ _Float16 As[NBUF][BM * SA];   // 3 x 20480 B
    __shared__ _Float16 Bs[NBUF][BN * SA];   // 3 x 10240 B

    const int tid  = threadIdx.x;
    const int lane = tid & 31;
    const int wave = tid >> 5;
    const int wm   = wave >> 1;   // 0..3 (M direction)
    const int wn   = wave & 1;    // 0..1 (N direction)
    const int half = lane >> 4;   // 0/1
    const int ln   = lane & 15;

    const int bn0 = blockIdx.x * BN;
    const int bm0 = blockIdx.y * BM;

    // staging maps: 16B chunks. A: 256 rows x 4 = 1024 (4/thr); B: 512 (2/thr)
    uint64_t aG[4]; uint32_t aL[4];
    uint64_t bG[2]; uint32_t bL[2];
    uint32_t ldsA[NBUF], ldsB[NBUF];
#pragma unroll
    for (int p = 0; p < NBUF; ++p) {
        ldsA[p] = (uint32_t)(uintptr_t)&As[p][0];
        ldsB[p] = (uint32_t)(uintptr_t)&Bs[p][0];
    }
#pragma unroll
    for (int j = 0; j < 4; ++j) {
        const int c = tid + 256 * j, r = c >> 2, q = c & 3;
        aL[j] = (uint32_t)(r * (SA * 2) + q * 16);
        aG[j] = (uint64_t)(uintptr_t)(Aq + (size_t)(bm0 + r) * IN_F + q * 8);
    }
#pragma unroll
    for (int j = 0; j < 2; ++j) {
        const int c = tid + 256 * j, r = c >> 2, q = c & 3;
        bL[j] = (uint32_t)(r * (SA * 2) + q * 16);
        bG[j] = (uint64_t)(uintptr_t)(Bw + (size_t)(bn0 + r) * IN_F + q * 8);
    }

    v8f acc[4][4];
#pragma unroll
    for (int mi = 0; mi < 4; ++mi)
#pragma unroll
        for (int ni = 0; ni < 4; ++ni)
            acc[mi][ni] = (v8f)(0.0f);

    // one K-step of fragment loads + 16 WMMAs from buffer `cur`
    auto compute_step = [&](int cur) {
        const int aoff = half * 8;
        v16h af[4];
#pragma unroll
        for (int mi = 0; mi < 4; ++mi) {
            const int r = wm * 64 + mi * 16 + ln;
            v8h lo = *(const v8h*)&As[cur][r * SA + aoff];
            v8h hi = *(const v8h*)&As[cur][r * SA + aoff + 16];
            af[mi] = __builtin_shufflevector(lo, hi, 0,1,2,3,4,5,6,7,8,9,10,11,12,13,14,15);
        }
        const int boff = half * 8;
        v16h bf[4];
#pragma unroll
        for (int ni = 0; ni < 4; ++ni) {
            const int r = wn * 64 + ni * 16 + ln;
            v8h lo = *(const v8h*)&Bs[cur][r * SA + boff];
            v8h hi = *(const v8h*)&Bs[cur][r * SA + boff + 16];
            bf[ni] = __builtin_shufflevector(lo, hi, 0,1,2,3,4,5,6,7,8,9,10,11,12,13,14,15);
        }
#pragma unroll
        for (int mi = 0; mi < 4; ++mi)
#pragma unroll
            for (int ni = 0; ni < 4; ++ni)
                acc[mi][ni] = __builtin_amdgcn_wmma_f32_16x16x32_f16(
                    false, af[mi], false, bf[ni], (short)0, acc[mi][ni], false, false);
    };

    // prologue: stage steps 0 and 1 into buffers 0 and 1
    async_stage(ldsA[0], ldsB[0], aL, aG, bL, bG);
    async_stage(ldsA[1], ldsB[1], aL, aG, bL, bG);
    asm volatile("s_wait_asynccnt 0x6" ::: "memory");   // buffer 0 landed
    __syncthreads();

    // steady state: s = 0 .. NSTEP-3, stage batch s+2, branchless wait 0x6
    int cur = 0, stage = 2;
    for (int s = 0; s < NSTEP - 2; ++s) {
        async_stage(ldsA[stage], ldsB[stage], aL, aG, bL, bG);
        compute_step(cur);
        asm volatile("s_wait_asynccnt 0x6"
                     :: "v"(acc[0][0][0]), "v"(acc[1][1][0]),
                        "v"(acc[2][2][0]), "v"(acc[3][3][0]) : "memory");
        __syncthreads();
        cur = (cur + 1 == NBUF) ? 0 : cur + 1;
        stage = (stage + 1 == NBUF) ? 0 : stage + 1;
    }

    // peeled step NSTEP-2: drain the final batch, then barrier
    compute_step(cur);
    asm volatile("s_wait_asynccnt 0x0"
                 :: "v"(acc[0][0][0]), "v"(acc[1][1][0]),
                    "v"(acc[2][2][0]), "v"(acc[3][3][0]) : "memory");
    __syncthreads();
    cur = (cur + 1 == NBUF) ? 0 : cur + 1;

    // peeled step NSTEP-1: nothing left to stage or wait for
    compute_step(cur);

    // epilogue: lanes 0..15 -> (M=r, N=ln); lanes 16..31 -> (M=r+8, N=ln)
#pragma unroll
    for (int ni = 0; ni < 4; ++ni) {
        const int gcol = bn0 + wn * 64 + ni * 16 + ln;
        const float be = beta[gcol] * (1.0f / QB);
        const float bi = bias[gcol];
#pragma unroll
        for (int mi = 0; mi < 4; ++mi) {
#pragma unroll
            for (int r = 0; r < 8; ++r) {
                const int grow = bm0 + wm * 64 + mi * 16 + r + half * 8;
                out[(size_t)grow * OUT_F + gcol] =
                    acc[mi][ni][r] * (be * gama[grow]) + bi;
            }
        }
    }
}

// ----------------------------------------------------------------------------
// Host-side launcher
// ----------------------------------------------------------------------------
extern "C" void kernel_launch(void* const* d_in, const int* in_sizes, int n_in,
                              void* d_out, int out_size, void* d_ws, size_t ws_size,
                              hipStream_t stream) {
    (void)in_sizes; (void)n_in; (void)out_size; (void)ws_size;

    const float* x    = (const float*)d_in[0];   // [BATCH, IN_F]
    const float* w    = (const float*)d_in[1];   // [OUT_F, IN_F]
    const float* bias = (const float*)d_in[2];   // [OUT_F]
    float* out        = (float*)d_out;           // [BATCH, OUT_F]

    char* ws = (char*)d_ws;
    const size_t xq_bytes = (size_t)BATCH * IN_F * sizeof(_Float16);   // 32 MB
    const size_t wb_bytes = (size_t)OUT_F * IN_F * sizeof(_Float16);   //  8 MB
    _Float16* xq   = (_Float16*)ws;
    _Float16* wbin = (_Float16*)(ws + xq_bytes);
    float*    gama = (float*)(ws + xq_bytes + wb_bytes);               // [BATCH]
    float*    beta = gama + BATCH;                                     // [OUT_F]

    prep_weights_kernel<<<OUT_F, 256, 0, stream>>>(w, wbin, beta);
    prep_acts_kernel<<<BATCH, 256, 0, stream>>>(x, xq, gama);

    dim3 grid(OUT_F / BN, BATCH / BM);   // (16, 32)
    bitlinear_gemm_kernel<<<grid, 256, 0, stream>>>(xq, wbin, gama, beta, bias, out);
}
